// CAA_Stable_40183714021508
// MI455X (gfx1250) — matrix operand order repaired and checked by reading, not compiled, not run on hardware
//
#include <hip/hip_runtime.h>
#include <hip/hip_bf16.h>

// ---------------------------------------------------------------------------
// Spatial self-attention (flash-attention style) for gfx1250 / MI455X.
// B=8, C=256, N=4096, D=16.  All GEMMs via v_wmma_f32_16x16x32_f16.
// ---------------------------------------------------------------------------

typedef __attribute__((ext_vector_type(16))) _Float16 v16h;
typedef __attribute__((ext_vector_type(8)))  _Float16 v8h;
typedef __attribute__((ext_vector_type(8)))  float    v8f;

#define BATCH 8
#define CCH   256
#define NN    4096
#define DD    16

static __device__ __forceinline__ v8f wmma_f16(v16h a, v16h b, v8f c) {
    return __builtin_amdgcn_wmma_f32_16x16x32_f16(false, a, false, b, (short)0, c,
                                                  false, false);
}

// A-operand (16-bit, 16x32): lane m = l&15; elements [0..15] = K base..base+15,
// base = (l>=16)*16.  Caller passes p pointing at
// row m, column (K) = base; second half is at p+8.
static __device__ __forceinline__ v16h load_a16(const _Float16* p) {
    v8h lo = *(const v8h*)p;
    v8h hi = *(const v8h*)(p + 8);
    v16h r;
#pragma unroll
    for (int i = 0; i < 8; ++i) { r[i] = lo[i]; r[8 + i] = hi[i]; }
    return r;
}

// ---------------------------------------------------------------------------
// 1) Weight conversion: Wq/Wk stacked -> Wqk f16 [32,256]; Wv,Wo -> f16.
// ---------------------------------------------------------------------------
__global__ void caa_convert_w(const float* __restrict__ Wq,
                              const float* __restrict__ Wk,
                              const float* __restrict__ Wv,
                              const float* __restrict__ Wo,
                              _Float16* __restrict__ Wqk,
                              _Float16* __restrict__ Wvh,
                              _Float16* __restrict__ Woh) {
    int i = blockIdx.x * blockDim.x + threadIdx.x;
    const int nqk = DD * CCH;           // 4096
    const int ncc = CCH * CCH;          // 65536
    if (i < nqk) {
        Wqk[i] = (_Float16)Wq[i];
    } else if (i < 2 * nqk) {
        int j = i - nqk;
        Wqk[nqk + j] = (_Float16)Wk[j];
    } else if (i < 2 * nqk + ncc) {
        int j = i - 2 * nqk;
        Wvh[j] = (_Float16)Wv[j];
    } else if (i < 2 * nqk + 2 * ncc) {
        int j = i - 2 * nqk - ncc;
        Woh[j] = (_Float16)Wo[j];
    }
}

// ---------------------------------------------------------------------------
// 2) LDS-tiled transpose+downconvert: x f32 [B,C,N] -> xT f16 [B,N,C].
// ---------------------------------------------------------------------------
__global__ __launch_bounds__(256) void caa_transpose_x(const float* __restrict__ x,
                                                       _Float16* __restrict__ xT) {
    __shared__ float tile[32][33];
    const int b  = blockIdx.z;
    const int c0 = blockIdx.y * 32;
    const int n0 = blockIdx.x * 32;
    const int tx = threadIdx.x;        // 0..31
    const int ty = threadIdx.y;        // 0..7
#pragma unroll
    for (int i = 0; i < 32; i += 8)
        tile[ty + i][tx] = x[((size_t)b * CCH + c0 + ty + i) * NN + n0 + tx];
    __syncthreads();
#pragma unroll
    for (int i = 0; i < 32; i += 8)
        xT[((size_t)b * NN + n0 + ty + i) * CCH + c0 + tx] = (_Float16)tile[tx][ty + i];
}

// ---------------------------------------------------------------------------
// 3a) q/k projection: Wqk[32,256] @ xT^T -> q,k f16 [B,N,16]  (one wave = one
//     16-query tile, two output tiles q & k, K-loop of 8 WMMAs each).
// ---------------------------------------------------------------------------
__global__ __launch_bounds__(128) void caa_proj_qk(const _Float16* __restrict__ xT,
                                                   const _Float16* __restrict__ Wqk,
                                                   const float* __restrict__ bq,
                                                   const float* __restrict__ bk,
                                                   _Float16* __restrict__ qo,
                                                   _Float16* __restrict__ ko) {
    const int lane = threadIdx.x & 31;
    const int wave = threadIdx.x >> 5;
    const int b    = blockIdx.y;
    const int n0   = (blockIdx.x * 4 + wave) * 16;
    const int half = lane >> 4;
    const int l15  = lane & 15;

    const _Float16* xrow = xT + ((size_t)b * NN + n0 + l15) * CCH + half * 16;
    const _Float16* wq   = Wqk + (size_t)l15 * CCH + half * 16;
    const _Float16* wk   = Wqk + (size_t)(16 + l15) * CCH + half * 16;

    v8f accq = {}, acck = {};
#pragma unroll
    for (int k0 = 0; k0 < CCH; k0 += 32) {
        v16h bx = *(const v16h*)(xrow + k0);   // B op: col n, K contiguous
        accq = wmma_f16(load_a16(wq + k0), bx, accq);
        acck = wmma_f16(load_a16(wk + k0), bx, acck);
    }
    // C/D tile: vgpr r, lane -> row d = r + half*8, col n = n0+l15.
    v8h qv, kv;
#pragma unroll
    for (int r = 0; r < 8; ++r) {
        qv[r] = (_Float16)(accq[r] + bq[half * 8 + r]);
        kv[r] = (_Float16)(acck[r] + bk[half * 8 + r]);
    }
    *(v8h*)(qo + ((size_t)b * NN + n0 + l15) * DD + half * 8) = qv;
    *(v8h*)(ko + ((size_t)b * NN + n0 + l15) * DD + half * 8) = kv;
}

// ---------------------------------------------------------------------------
// 3b) v projection: Wv[256,256] @ x -> v f16 [B,C,N] (keys contiguous).
// ---------------------------------------------------------------------------
__global__ __launch_bounds__(128) void caa_proj_v(const _Float16* __restrict__ xT,
                                                  const _Float16* __restrict__ Wvh,
                                                  const float* __restrict__ bv,
                                                  _Float16* __restrict__ vo) {
    const int lane = threadIdx.x & 31;
    const int wave = threadIdx.x >> 5;
    const int b    = blockIdx.z;
    const int c0   = blockIdx.y * 16;
    const int n0   = (blockIdx.x * 4 + wave) * 16;
    const int half = lane >> 4;
    const int l15  = lane & 15;

    const _Float16* xrow = xT + ((size_t)b * NN + n0 + l15) * CCH + half * 16;
    const _Float16* wrow = Wvh + (size_t)(c0 + l15) * CCH + half * 16;

    v8f acc = {};
#pragma unroll
    for (int k0 = 0; k0 < CCH; k0 += 32)
        acc = wmma_f16(load_a16(wrow + k0), *(const v16h*)(xrow + k0), acc);

#pragma unroll
    for (int r = 0; r < 8; ++r) {
        int c = c0 + r + half * 8;
        vo[((size_t)b * CCH + c) * NN + n0 + l15] = (_Float16)(acc[r] + bv[c]);
    }
}

// ---------------------------------------------------------------------------
// 4) Flash attention: per wave, 16 queries x full C=256.  Online softmax with
//    clip(+-50) and scale 0.25.  acc = 16 f32 tiles (128 VGPRs).
// ---------------------------------------------------------------------------
__global__ __launch_bounds__(256, 1) void caa_attention(const _Float16* __restrict__ q,
                                                        const _Float16* __restrict__ k,
                                                        const _Float16* __restrict__ v,
                                                        _Float16* __restrict__ O) {
    __shared__ _Float16 lds[8][16][32];    // per-wave P staging (1 KB each)
    const int lane = threadIdx.x & 31;
    const int wave = threadIdx.x >> 5;
    const int b    = blockIdx.y;
    const int q0   = blockIdx.x * 128 + wave * 16;
    const int half = lane >> 4;
    const int l15  = lane & 15;

    // Q A-operand: D=16 padded to K=32 with zeros.
    v16h qa;
    if (half == 0) {
        qa = *(const v16h*)(q + ((size_t)b * NN + q0 + l15) * DD);
    } else {
        qa = (v16h){};
    }

    const _Float16* kb = k + (size_t)b * NN * DD;
    const _Float16* vb = v + (size_t)b * CCH * NN;

    v8f acc[16];
#pragma unroll
    for (int t = 0; t < 16; ++t) acc[t] = (v8f){};
    float m_r[8], l_r[8];
#pragma unroll
    for (int r = 0; r < 8; ++r) { m_r[r] = -1e30f; l_r[r] = 0.f; }

    const v8f zero8 = {};
    for (int n0 = 0; n0 < NN; n0 += 32) {
        // K B-operands: lanes 0-15 carry K(=D)0..15 (real), lanes 16-31 K16..31 = 0 pad.
        v16h kb0 = {}, kb1 = {};
        if (half == 0) {
            kb0 = *(const v16h*)(kb + (size_t)(n0 + l15) * DD);
            kb1 = *(const v16h*)(kb + (size_t)(n0 + 16 + l15) * DD);
        }
        v8f s0 = wmma_f16(qa, kb0, zero8);
        v8f s1 = wmma_f16(qa, kb1, zero8);

        float mnew[8];
#pragma unroll
        for (int r = 0; r < 8; ++r) {
            float a0 = fminf(fmaxf(s0[r] * 0.25f, -50.f), 50.f);
            float a1 = fminf(fmaxf(s1[r] * 0.25f, -50.f), 50.f);
            s0[r] = a0; s1[r] = a1;
            float mx = fmaxf(a0, a1);
            mx = fmaxf(mx, __shfl_xor(mx, 1, 32));
            mx = fmaxf(mx, __shfl_xor(mx, 2, 32));
            mx = fmaxf(mx, __shfl_xor(mx, 4, 32));
            mx = fmaxf(mx, __shfl_xor(mx, 8, 32));
            mnew[r] = fmaxf(m_r[r], mx);
        }
#pragma unroll
        for (int r = 0; r < 8; ++r) {
            float alpha = __expf(m_r[r] - mnew[r]);
            float p0 = __expf(s0[r] - mnew[r]);
            float p1 = __expf(s1[r] - mnew[r]);
            s0[r] = p0; s1[r] = p1;
            float rs = p0 + p1;
            rs += __shfl_xor(rs, 1, 32);
            rs += __shfl_xor(rs, 2, 32);
            rs += __shfl_xor(rs, 4, 32);
            rs += __shfl_xor(rs, 8, 32);
            l_r[r] = l_r[r] * alpha + rs;
            m_r[r] = mnew[r];
#pragma unroll
            for (int t = 0; t < 16; ++t) acc[t][r] *= alpha;
        }

        // C-layout -> A-layout relayout through per-wave LDS.
#pragma unroll
        for (int r = 0; r < 8; ++r) {
            int row = r + half * 8;
            lds[wave][row][l15]      = (_Float16)s0[r];
            lds[wave][row][16 + l15] = (_Float16)s1[r];
        }
        asm volatile("s_wait_dscnt 0" ::: "memory");
        v16h pa;
        {
            const _Float16* lp = &lds[wave][l15][half * 16];
            v8h lo = *(const v8h*)lp;
            v8h hi = *(const v8h*)(lp + 8);
#pragma unroll
            for (int i = 0; i < 8; ++i) { pa[i] = lo[i]; pa[8 + i] = hi[i]; }
        }
        asm volatile("s_wait_dscnt 0" ::: "memory");

        // P (16x32) @ V (32 keys x 16 ch) over 16 channel tiles.
#pragma unroll
        for (int t = 0; t < 16; ++t) {
            v16h vbop = *(const v16h*)(vb + (size_t)(t * 16 + l15) * NN + n0 + half * 16);
            acc[t] = wmma_f16(pa, vbop, acc[t]);
        }
    }

    float inv[8];
#pragma unroll
    for (int r = 0; r < 8; ++r) inv[r] = 1.0f / l_r[r];
    _Float16* Ob = O + ((size_t)b * NN + q0) * CCH;
#pragma unroll
    for (int t = 0; t < 16; ++t)
#pragma unroll
        for (int r = 0; r < 8; ++r)
            Ob[(size_t)(r + half * 8) * CCH + t * 16 + l15] =
                (_Float16)(acc[t][r] * inv[r]);
}

// ---------------------------------------------------------------------------
// 5) Output conv + clamped-gamma residual: y = g*(Wo@O^T + bo) + x   (f32 out)
// ---------------------------------------------------------------------------
__global__ __launch_bounds__(256) void caa_outconv(const _Float16* __restrict__ O,
                                                   const _Float16* __restrict__ Woh,
                                                   const float* __restrict__ bo,
                                                   const float* __restrict__ gamma,
                                                   const float* __restrict__ x,
                                                   float* __restrict__ y) {
    const int lane = threadIdx.x & 31;
    const int wave = threadIdx.x >> 5;
    const int b    = blockIdx.z;
    const int f0   = blockIdx.y * 16;
    const int n0   = (blockIdx.x * 8 + wave) * 16;
    const int half = lane >> 4;
    const int l15  = lane & 15;

    const _Float16* orow = O + ((size_t)b * NN + n0 + l15) * CCH + half * 16;
    const _Float16* wrow = Woh + (size_t)(f0 + l15) * CCH + half * 16;

    v8f acc = {};
#pragma unroll
    for (int k0 = 0; k0 < CCH; k0 += 32)
        acc = wmma_f16(load_a16(wrow + k0), *(const v16h*)(orow + k0), acc);

    float g = fminf(fmaxf(gamma[0], 0.f), 1.f);
#pragma unroll
    for (int r = 0; r < 8; ++r) {
        int f = f0 + r + half * 8;
        size_t idx = ((size_t)b * CCH + f) * NN + n0 + l15;
        y[idx] = g * (acc[r] + bo[f]) + x[idx];
    }
}

// ---------------------------------------------------------------------------
extern "C" void kernel_launch(void* const* d_in, const int* in_sizes, int n_in,
                              void* d_out, int out_size, void* d_ws, size_t ws_size,
                              hipStream_t stream) {
    const float* x     = (const float*)d_in[0];
    const float* Wq    = (const float*)d_in[1];
    const float* bq    = (const float*)d_in[2];
    const float* Wk    = (const float*)d_in[3];
    const float* bk    = (const float*)d_in[4];
    const float* Wv    = (const float*)d_in[5];
    const float* bv    = (const float*)d_in[6];
    const float* Wo    = (const float*)d_in[7];
    const float* bo    = (const float*)d_in[8];
    const float* gamma = (const float*)d_in[9];
    float* y = (float*)d_out;

    char* ws = (char*)d_ws;
    size_t off = 0;
    auto take = [&](size_t bytes) {
        size_t o = off;
        off = (off + bytes + 255) & ~(size_t)255;
        return o;
    };
    _Float16* xT  = (_Float16*)(ws + take((size_t)BATCH * NN * CCH * 2)); // 16 MB
    _Float16* Wqk = (_Float16*)(ws + take((size_t)32 * CCH * 2));
    _Float16* Wvh = (_Float16*)(ws + take((size_t)CCH * CCH * 2));
    _Float16* Woh = (_Float16*)(ws + take((size_t)CCH * CCH * 2));
    _Float16* qb  = (_Float16*)(ws + take((size_t)BATCH * NN * DD * 2));  // 1 MB
    _Float16* kbf = (_Float16*)(ws + take((size_t)BATCH * NN * DD * 2));  // 1 MB
    _Float16* vbf = (_Float16*)(ws + take((size_t)BATCH * CCH * NN * 2)); // 16 MB
    _Float16* Ob  = (_Float16*)(ws + take((size_t)BATCH * NN * CCH * 2)); // 16 MB
    (void)ws_size; (void)in_sizes; (void)n_in; (void)out_size;

    // 1) weights -> f16
    {
        int total = 2 * DD * CCH + 2 * CCH * CCH;
        caa_convert_w<<<(total + 255) / 256, 256, 0, stream>>>(Wq, Wk, Wv, Wo,
                                                               Wqk, Wvh, Woh);
    }
    // 2) x -> xT f16 [B,N,C]
    caa_transpose_x<<<dim3(NN / 32, CCH / 32, BATCH), dim3(32, 8), 0, stream>>>(x, xT);
    // 3) projections
    caa_proj_qk<<<dim3(NN / 64, BATCH), 128, 0, stream>>>(xT, Wqk, bq, bk, qb, kbf);
    caa_proj_v<<<dim3(NN / 64, CCH / 16, BATCH), 128, 0, stream>>>(xT, Wvh, bv, vbf);
    // 4) flash attention
    caa_attention<<<dim3(NN / 128, BATCH), 256, 0, stream>>>(qb, kbf, vbf, Ob);
    // 5) output conv + residual
    caa_outconv<<<dim3(NN / 128, CCH / 16, BATCH), 256, 0, stream>>>(Ob, Woh, bo,
                                                                     gamma, x, y);
}
